// Model_23940147708540
// MI455X (gfx1250) — matrix operand order repaired and checked by reading, not compiled, not run on hardware
//
#include <hip/hip_runtime.h>
#include <stdint.h>

#define B_ 4
#define H_ 3
#define S_ 1536
#define D_ 768

typedef __attribute__((ext_vector_type(16))) __bf16   v16bf;
typedef __attribute__((ext_vector_type(2)))  __bf16   v2bf;
typedef __attribute__((ext_vector_type(8)))  float    v8f;
typedef __attribute__((ext_vector_type(4)))  uint32_t v4u;
typedef __attribute__((ext_vector_type(2)))  float    v2f;
typedef __attribute__((ext_vector_type(4)))  float    v4f;

union ABFrag {
    v4u      q[2];
    uint32_t u[8];
    v16bf    v;
};

// Pack two f32 -> packed bf16 pair (lo in [15:0], hi in [31:16]).
__device__ __forceinline__ uint32_t cvt_pk(float lo, float hi) {
#if __has_builtin(__builtin_amdgcn_cvt_pk_bf16_f32)
    auto r = __builtin_amdgcn_cvt_pk_bf16_f32(lo, hi);
    return __builtin_bit_cast(uint32_t, r);
#else
    v2bf t;
    t[0] = (__bf16)lo;
    t[1] = (__bf16)hi;
    return __builtin_bit_cast(uint32_t, t);
#endif
}

__device__ __forceinline__ void async_ld_b128(uint32_t lds_addr, uint64_t gaddr) {
    asm volatile("global_load_async_to_lds_b128 %0, %1, off"
                 :: "v"(lds_addr), "v"(gaddr) : "memory");
}

template <int N>
__device__ __forceinline__ void wait_async() {
#if __has_builtin(__builtin_amdgcn_s_wait_asynccnt)
    __builtin_amdgcn_s_wait_asynccnt(N);
#else
    asm volatile("s_wait_asynccnt %0" :: "i"(N) : "memory");
#endif
}

// Issue 24 async b128 loads covering one contiguous 96 KB V chunk.
__device__ __forceinline__ void stage_chunk(const float* gsrc, const void* ldst, int tid) {
    const uint64_t g = (uint64_t)(uintptr_t)gsrc;
    const uint32_t l = (uint32_t)(uintptr_t)ldst;
    #pragma unroll
    for (int i = 0; i < 24; ++i) {
        uint32_t off = (uint32_t)(tid + 256 * i) * 16u;
        async_ld_b128(l + off, g + off);
    }
}

// LDS layout (dynamic, 288 KB total):
//   [0      , 98304 )  Sc : f32 scores [16][1536] (ph 2-3); V chunk buf0 (ph 4)
//   [98304  , 147456)  P  : bf16 probs [16][1536] (ph 3-4)
//   [147456 , 196608)  Qs : bf16 Q tile [16][768] (ph 1-2)
//   [196608 , 294912)  V chunk buf1 [32][768] f32 (ph 4)
#define SC_OFF 0
#define P_OFF  98304
#define Q_OFF  147456
#define V2_OFF 196608
#define SMEM_BYTES 294912

__global__ __launch_bounds__(256) void attn_fused_kernel(
    const float* __restrict__ Q, const float* __restrict__ K,
    const float* __restrict__ V, const float* __restrict__ mask,
    const float* __restrict__ U, float* __restrict__ out)
{
    extern __shared__ char smem[];
    float*    Sc = (float*)(smem + SC_OFF);
    uint16_t* P  = (uint16_t*)(smem + P_OFF);
    char*     QS = smem + Q_OFF;

    const int tid  = threadIdx.x;
    const int wave = tid >> 5;
    const int lane = tid & 31;
    const int lm   = lane & 15;   // N index (B/C/D) or M index (A)
    const int lh   = lane >> 4;   // lane half

    const int qt    = blockIdx.x % (S_ / 16);
    const int bh    = blockIdx.x / (S_ / 16);
    const int qbase = qt * 16;

    const float* Qg = Q + (size_t)bh * S_ * D_ + (size_t)qbase * D_;
    const float* Kg = K + (size_t)bh * S_ * D_;
    const float* Vg = V + (size_t)bh * S_ * D_;
    const float* Ug = U + (size_t)bh * S_ * S_ + (size_t)qbase * S_;
    float*       Og = out + (size_t)bh * S_ * D_ + (size_t)qbase * D_;

    // ---------------- Phase 1: Q tile -> LDS bf16 [16][768] ----------------
    {
        uint32_t* Qw = (uint32_t*)QS;
        #pragma unroll
        for (int i = 0; i < 24; ++i) {
            int p = tid + 256 * i;          // 0..6143 packed-pair index
            int m = p / 384;                // row 0..15
            int j = p - m * 384;            // pair col 0..383
            v2f q2 = *(const v2f*)(Qg + (size_t)m * D_ + 2 * j);
            Qw[p] = cvt_pk(q2[0], q2[1]);
        }
    }
    __syncthreads();

    const float scale = 0.03608439182435161f;   // 1/sqrt(768)

    // ------- Phase 2: S = Q K^T * scale + mask -> LDS f32 -------
    // A fragment loaded once per d-chunk; 12 acc tiles live. B fragments
    // built in simultaneously-live pairs so RA cannot coalesce them and the
    // two WMMAs issue back-to-back without WAR NOP gaps.
    {
        v8f sacc[12];
        #pragma unroll
        for (int t = 0; t < 12; ++t) { v8f z = {}; sacc[t] = z; }

        for (int c = 0; c < 24; ++c) {
            ABFrag a;   // Q row lm, d = 32c + 16*lh .. (+15)
            const char* abase = QS + lm * (768 * 2) + (c * 32 + 16 * lh) * 2;
            a.q[0] = *(const v4u*)(abase);
            a.q[1] = *(const v4u*)(abase + 16);
            #pragma unroll
            for (int tp = 0; tp < 6; ++tp) {
                const int t0 = 2 * tp;
                const float* kp0 =
                    Kg + (size_t)((wave + 8 * t0) * 16 + lm) * D_ + c * 32 + 16 * lh;
                const float* kp1 = kp0 + (size_t)128 * D_;   // next tile: +8*16 rows
                ABFrag b0, b1;
                #pragma unroll
                for (int vv = 0; vv < 4; ++vv) {
                    v4f f0 = *(const v4f*)(kp0 + 4 * vv);
                    v4f f1 = *(const v4f*)(kp1 + 4 * vv);
                    b0.u[2 * vv]     = cvt_pk(f0[0], f0[1]);
                    b0.u[2 * vv + 1] = cvt_pk(f0[2], f0[3]);
                    b1.u[2 * vv]     = cvt_pk(f1[0], f1[1]);
                    b1.u[2 * vv + 1] = cvt_pk(f1[2], f1[3]);
                }
                sacc[t0] = __builtin_amdgcn_wmma_f32_16x16x32_bf16(
                    false, a.v, false, b0.v, (short)0, sacc[t0], false, false);
                sacc[t0 + 1] = __builtin_amdgcn_wmma_f32_16x16x32_bf16(
                    false, a.v, false, b1.v, (short)0, sacc[t0 + 1], false, false);
            }
        }
        #pragma unroll
        for (int t = 0; t < 12; ++t) {
            const int kb = (wave + 8 * t) * 16;
            #pragma unroll
            for (int r = 0; r < 8; ++r) {
                int m = r + 8 * lh;
                float s = sacc[t][r] * scale + mask[(size_t)(qbase + m) * S_ + kb + lm];
                Sc[m * S_ + kb + lm] = s;
            }
        }
    }
    __syncthreads();

    // ---------------- Phase 3: softmax + dropout -> P (bf16) ---------------
    #pragma unroll
    for (int rr = 0; rr < 2; ++rr) {
        const int row = 2 * wave + rr;
        float* sr = Sc + row * S_;
        float mx = -3.0e38f;
        for (int k = lane; k < S_; k += 32) mx = fmaxf(mx, sr[k]);
        #pragma unroll
        for (int o = 16; o > 0; o >>= 1) mx = fmaxf(mx, __shfl_xor(mx, o, 32));
        float sum = 0.f;
        for (int k = lane; k < S_; k += 32) {
            float e = __expf(sr[k] - mx);
            sr[k] = e;                       // cache exp in place
            sum += e;
        }
        #pragma unroll
        for (int o = 16; o > 0; o >>= 1) sum += __shfl_xor(sum, o, 32);
        const float inv = 1.0f / (sum * 0.9f);   // softmax norm * 1/(1-p_drop)
        const float* ur = Ug + (size_t)row * S_;
        uint16_t* pr = P + row * S_;
        for (int k = lane; k < S_; k += 32) {
            float p = (ur[k] > 0.1f) ? sr[k] * inv : 0.0f;
            v2bf t; t[0] = (__bf16)p; t[1] = (__bf16)0.f;
            pr[k] = (uint16_t)(__builtin_bit_cast(uint32_t, t) & 0xFFFFu);
        }
    }
    __syncthreads();

    // ---------------- Phase 4: O = P V (double-buffered async DMA) ---------
    v8f accv[6];
    #pragma unroll
    for (int j = 0; j < 6; ++j) { v8f z = {}; accv[j] = z; }

    // Prefetch chunk 0 into buf0 (= Sc region, dead after phase 3's barrier).
    stage_chunk(Vg, smem + SC_OFF, tid);

    for (int c = 0; c < 48; ++c) {
        if (c < 47) {
            // Issue next chunk's DMA into the other buffer (its previous
            // consumer finished at the end-of-iteration barrier of c-1).
            const char* nbuf = smem + (((c + 1) & 1) ? V2_OFF : SC_OFF);
            stage_chunk(Vg + (size_t)(c + 1) * 32 * D_, nbuf, tid);
            wait_async<24>();   // in-order: chunk c's 24 loads are complete
        } else {
            wait_async<0>();
        }
        __syncthreads();

        const float* Vf = (const float*)(smem + ((c & 1) ? V2_OFF : SC_OFF));

        // A fragment: P row lm, keys 32c + 16*lh .. (+15)
        ABFrag a;
        const char* abase = (const char*)P + lm * (S_ * 2) + (c * 32 + 16 * lh) * 2;
        a.q[0] = *(const v4u*)(abase);
        a.q[1] = *(const v4u*)(abase + 16);

        #pragma unroll
        for (int jp = 0; jp < 3; ++jp) {
            const int dcol0 = (wave * 6 + 2 * jp) * 16 + lm;
            const int dcol1 = dcol0 + 16;
            ABFrag b0, b1;   // V rows k = 2v + 16*lh (+1), columns dcol0/dcol1
            #pragma unroll
            for (int v = 0; v < 8; ++v) {
                const int k = 2 * v + 16 * lh;
                b0.u[v] = cvt_pk(Vf[k * D_ + dcol0], Vf[(k + 1) * D_ + dcol0]);
                b1.u[v] = cvt_pk(Vf[k * D_ + dcol1], Vf[(k + 1) * D_ + dcol1]);
            }
            accv[2 * jp] = __builtin_amdgcn_wmma_f32_16x16x32_bf16(
                false, a.v, false, b0.v, (short)0, accv[2 * jp], false, false);
            accv[2 * jp + 1] = __builtin_amdgcn_wmma_f32_16x16x32_bf16(
                false, a.v, false, b1.v, (short)0, accv[2 * jp + 1], false, false);
        }
        __syncthreads();
    }

    // ---------------- Epilogue: store O ------------------------------------
    #pragma unroll
    for (int j = 0; j < 6; ++j) {
        const int dcol = (wave * 6 + j) * 16 + lm;
        #pragma unroll
        for (int r = 0; r < 8; ++r) {
            int m = r + 8 * lh;
            Og[(size_t)m * D_ + dcol] = accv[j][r];
        }
    }
}

extern "C" void kernel_launch(void* const* d_in, const int* in_sizes, int n_in,
                              void* d_out, int out_size, void* d_ws, size_t ws_size,
                              hipStream_t stream) {
    (void)in_sizes; (void)n_in; (void)out_size; (void)d_ws; (void)ws_size;
    const float* Q    = (const float*)d_in[0];
    const float* K    = (const float*)d_in[1];
    const float* V    = (const float*)d_in[2];
    const float* mask = (const float*)d_in[3];
    const float* U    = (const float*)d_in[4];
    float* out = (float*)d_out;

    dim3 grid(B_ * H_ * (S_ / 16));
    dim3 block(256);
    attn_fused_kernel<<<grid, block, SMEM_BYTES, stream>>>(Q, K, V, mask, U, out);
}
